// DeepSeekRelationalModel_25443386261956
// MI455X (gfx1250) — compile-verified, matrix operand order checked by reading, not run on hardware
//
#include <hip/hip_runtime.h>
#include <hip/hip_bf16.h>

// ---------------------------------------------------------------------------
// DeepSeek-style MoE (top-2 of 16 experts, SwiGLU D=1024->H=512->D=1024)
// for gfx1250 (MI455X): f16 WMMA (v_wmma_f32_16x16x32_f16) with fp32 accum.
// T = B*S = 8192 tokens. Sparse (grouped-by-expert) execution:
//   router -> per-expert token lists -> per-expert tiled WMMA GEMMs.
// Round 3: prefetch locality 3 so global_prefetch pulls into WGP cache
// (round 2 lowered to SCOPE_SYS = GL2-only, which is useless for the
// L0-latency the inner loop actually eats).
// ---------------------------------------------------------------------------

#define MOE_D 1024
#define MOE_H 512
#define MOE_E 16
#define TILE_M 32

typedef __attribute__((ext_vector_type(16))) _Float16 v16h;
typedef __attribute__((ext_vector_type(8)))  float    v8f;

union Frag16 {           // one WMMA 16-bit operand fragment: 16 halves = 32B = 8 VGPRs
    v16h v;
    uint4 q[2];
};

#define WMMA_F16(A, B, C) \
    __builtin_amdgcn_wmma_f32_16x16x32_f16(false, (A).v, false, (B).v, (short)0, (C), false, false)

#define ZERO8 {0.f,0.f,0.f,0.f,0.f,0.f,0.f,0.f}

// -------------------------- f32 -> f16 convert (x) --------------------------
__global__ __launch_bounds__(256) void cvt_x_kernel(const float* __restrict__ in,
                                                    _Float16* __restrict__ out,
                                                    long long n8) {
    long long i = (long long)blockIdx.x * blockDim.x + threadIdx.x;
    if (i >= n8) return;
    long long base = i * 8;
    float4 a = *(const float4*)(in + base);
    float4 b = *(const float4*)(in + base + 4);
    union { _Float16 h[8]; uint4 q; } u;
    u.h[0] = (_Float16)a.x; u.h[1] = (_Float16)a.y;
    u.h[2] = (_Float16)a.z; u.h[3] = (_Float16)a.w;
    u.h[4] = (_Float16)b.x; u.h[5] = (_Float16)b.y;
    u.h[6] = (_Float16)b.z; u.h[7] = (_Float16)b.w;
    *(uint4*)(out + base) = u.q;
}

// ---------------- f32 -> f16 transpose-convert: [E][R][C] -> [E][C][R] -------
// Puts the GEMM K-dimension contiguous so B fragments are 2x b128 loads/lane.
__global__ __launch_bounds__(256) void transpose_cvt_kernel(const float* __restrict__ in,
                                                            _Float16* __restrict__ out,
                                                            int R, int C) {
    __shared__ float tile[32][33];
    const int e = blockIdx.z;
    const float* src = in + (size_t)e * R * C;
    _Float16* dst = out + (size_t)e * R * C;
    const int c0 = blockIdx.x * 32, r0 = blockIdx.y * 32;
    const int tx = threadIdx.x, ty = threadIdx.y;
#pragma unroll
    for (int j = 0; j < 4; ++j)
        tile[ty + 8 * j][tx] = src[(size_t)(r0 + ty + 8 * j) * C + c0 + tx];
    __syncthreads();
#pragma unroll
    for (int j = 0; j < 4; ++j)
        dst[(size_t)(c0 + ty + 8 * j) * R + r0 + tx] = (_Float16)tile[tx][ty + 8 * j];
}

// ------------------------------- router -------------------------------------
// One wave32 per token: lanes (e, half) split the D-dot-product, in-wave
// softmax over 16 experts, serial top-2, renormalize, append to expert lists.
__global__ __launch_bounds__(256) void router_kernel(const float* __restrict__ x,
                                                     const float* __restrict__ gw, // [D][E]
                                                     int* __restrict__ cnt,        // [E]
                                                     int* __restrict__ tok_list,   // [E][T]
                                                     float* __restrict__ wgt_list, // [E][T]
                                                     int T) {
    const int t = blockIdx.x * 8 + (threadIdx.x >> 5);
    if (t >= T) return;
    const int lane = threadIdx.x & 31;
    const int e = lane & 15;
    const int half = lane >> 4;

    float acc = 0.f;
    const float* xr = x + (size_t)t * MOE_D;
    for (int d = half; d < MOE_D; d += 2)
        acc = fmaf(xr[d], gw[d * MOE_E + e], acc);
    acc += __shfl_xor(acc, 16);          // now all 32 lanes hold score[lane&15]

    float m = acc;
#pragma unroll
    for (int msk = 1; msk <= 8; msk <<= 1) m = fmaxf(m, __shfl_xor(m, msk));
    float p = __expf(acc - m);
    float s = p;
#pragma unroll
    for (int msk = 1; msk <= 8; msk <<= 1) s += __shfl_xor(s, msk);
    const float prob = p / s;

    // serial top-2 over the 16 probabilities (lane ee < 16 holds expert ee)
    float b1 = -1.f, b2 = -1.f; int i1 = 0, i2 = 0;
#pragma unroll
    for (int ee = 0; ee < 16; ++ee) {
        float v = __shfl(prob, ee);
        if (v > b1)      { b2 = b1; i2 = i1; b1 = v; i1 = ee; }
        else if (v > b2) { b2 = v; i2 = ee; }
    }
    const float inv = 1.f / (b1 + b2);
    if (lane == 0) {
        int s1 = atomicAdd(&cnt[i1], 1);
        tok_list[(size_t)i1 * T + s1] = t;
        wgt_list[(size_t)i1 * T + s1] = b1 * inv;
        int s2 = atomicAdd(&cnt[i2], 1);
        tok_list[(size_t)i2 * T + s2] = t;
        wgt_list[(size_t)i2 * T + s2] = b2 * inv;
    }
}

// --------------------------- expert SwiGLU GEMMs -----------------------------
// One workgroup = 16 waves per (expert, 32-token tile), two 16-row m-tiles.
// Phase A: h[32,512] = silu(x@w1) * (x@w3)   (each wave: 32 H-columns)
// Phase B: out += wgt * (h @ w2)             (each wave: 64 D-columns)
__global__ __launch_bounds__(512) void moe_expert_kernel(
        const _Float16* __restrict__ xh,   // [T][D]
        const _Float16* __restrict__ w1t,  // [E][H][D]  (K contiguous)
        const _Float16* __restrict__ w3t,  // [E][H][D]
        const _Float16* __restrict__ w2t,  // [E][D][H]  (K contiguous)
        const int* __restrict__ cnt,
        const int* __restrict__ tok_list,  // [E][T]
        const float* __restrict__ wgt_list,
        float* __restrict__ out,           // [T][D]
        int T, int tiles_per_e) {
    const int e  = blockIdx.x / tiles_per_e;
    const int mt = blockIdx.x % tiles_per_e;
    const int n_e = cnt[e];
    if (mt * TILE_M >= n_e) return;        // block-uniform early exit

    __shared__ __align__(16) _Float16 xs[TILE_M][MOE_D];  // 64 KB
    __shared__ __align__(16) _Float16 hs[TILE_M][MOE_H];  // 32 KB
    __shared__ int   toks[TILE_M];
    __shared__ float wgts[TILE_M];

    const int tid = threadIdx.x;
    if (tid < TILE_M) {
        int gm = mt * TILE_M + tid;
        if (gm < n_e) {
            toks[tid] = tok_list[(size_t)e * T + gm];
            wgts[tid] = wgt_list[(size_t)e * T + gm];
        } else {                            // pad row: any valid token, weight 0
            toks[tid] = tok_list[(size_t)e * T + mt * TILE_M];
            wgts[tid] = 0.f;
        }
    }
    __syncthreads();

    // gather 32 token rows of x (f16) into LDS, 128B per thread, coalesced
    {
        const int r = tid >> 4, c = tid & 15;          // 16 threads per 2KB row
        const uint4* src = (const uint4*)(xh + (size_t)toks[r] * MOE_D);
        uint4* dstq = (uint4*)(&xs[r][0]);
#pragma unroll
        for (int j = 0; j < 8; ++j) dstq[c + 16 * j] = src[c + 16 * j];
    }
    __syncthreads();

    const int lane = tid & 31;
    const int wave = tid >> 5;             // 0..15
    const int nlo = lane & 15;             // row (A) / col (B,C) within tile
    const int hi  = lane >> 4;             // K-interleave select

    // ---------------- Phase A: gate/up projections over K = D -----------------
    // acc[g|u][m-tile][n-tile]
    v8f ag00 = ZERO8, ag01 = ZERO8, ag10 = ZERO8, ag11 = ZERO8;
    v8f au00 = ZERO8, au01 = ZERO8, au10 = ZERO8, au11 = ZERO8;
    {
        const _Float16* w1e = w1t + (size_t)e * MOE_H * MOE_D;
        const _Float16* w3e = w3t + (size_t)e * MOE_H * MOE_D;
        const int colA = wave * 32 + nlo;              // n-tile 0 column
        const _Float16* p1a = w1e + (size_t)colA * MOE_D + hi * 8;
        const _Float16* p1b = p1a + (size_t)16 * MOE_D;
        const _Float16* p3a = w3e + (size_t)colA * MOE_D + hi * 8;
        const _Float16* p3b = p3a + (size_t)16 * MOE_D;

        for (int k0 = 0; k0 < MOE_D; k0 += 32) {
            // batch ALL fragment loads for this k-step (distinct registers)
            Frag16 a0, a1, bg0, bg1, bu0, bu1;
            a0.q[0] = *(const uint4*)(&xs[nlo][k0 + hi * 8]);
            a0.q[1] = *(const uint4*)(&xs[nlo][k0 + 16 + hi * 8]);
            a1.q[0] = *(const uint4*)(&xs[16 + nlo][k0 + hi * 8]);
            a1.q[1] = *(const uint4*)(&xs[16 + nlo][k0 + 16 + hi * 8]);
            bg0.q[0] = *(const uint4*)(p1a + k0);
            bg0.q[1] = *(const uint4*)(p1a + k0 + 16);
            bg1.q[0] = *(const uint4*)(p1b + k0);
            bg1.q[1] = *(const uint4*)(p1b + k0 + 16);
            bu0.q[0] = *(const uint4*)(p3a + k0);
            bu0.q[1] = *(const uint4*)(p3a + k0 + 16);
            bu1.q[0] = *(const uint4*)(p3b + k0);
            bu1.q[1] = *(const uint4*)(p3b + k0 + 16);
            // warm the WGP cache ~16 k-steps ahead (global_prefetch_b8)
            __builtin_prefetch((const void*)(p1a + k0 + 512), 0, 3);
            __builtin_prefetch((const void*)(p3a + k0 + 512), 0, 3);
            // 8 independent WMMAs, each B fragment feeds two m-tiles
            ag00 = WMMA_F16(a0, bg0, ag00);
            ag10 = WMMA_F16(a1, bg0, ag10);
            ag01 = WMMA_F16(a0, bg1, ag01);
            ag11 = WMMA_F16(a1, bg1, ag11);
            au00 = WMMA_F16(a0, bu0, au00);
            au10 = WMMA_F16(a1, bu0, au10);
            au01 = WMMA_F16(a0, bu1, au01);
            au11 = WMMA_F16(a1, bu1, au11);
        }
    }
    // fused SiLU(g)*u -> f16 h-tile in LDS (C layout: lane->(M=j+8*hi, N=nlo))
    {
        const int colA = wave * 32 + nlo;
#pragma unroll
        for (int j = 0; j < 8; ++j) {
            const int r = j + 8 * hi;
            float g, u;
            g = ag00[j]; u = au00[j];
            hs[r][colA] = (_Float16)((g / (1.f + __expf(-g))) * u);
            g = ag01[j]; u = au01[j];
            hs[r][colA + 16] = (_Float16)((g / (1.f + __expf(-g))) * u);
            g = ag10[j]; u = au10[j];
            hs[r + 16][colA] = (_Float16)((g / (1.f + __expf(-g))) * u);
            g = ag11[j]; u = au11[j];
            hs[r + 16][colA + 16] = (_Float16)((g / (1.f + __expf(-g))) * u);
        }
    }
    __syncthreads();

    // ---------------- Phase B: down projection over K = H ---------------------
    // acc[m-tile][n-tile], waves own 64 D-columns (4 n-tiles)
    v8f c00 = ZERO8, c01 = ZERO8, c02 = ZERO8, c03 = ZERO8;
    v8f c10 = ZERO8, c11 = ZERO8, c12 = ZERO8, c13 = ZERO8;
    {
        const _Float16* w2e = w2t + (size_t)e * MOE_D * MOE_H;
        const int colB = wave * 64 + nlo;
        const _Float16* p0 = w2e + (size_t)colB * MOE_H + hi * 8;
        const _Float16* p1 = p0 + (size_t)16 * MOE_H;
        const _Float16* p2 = p0 + (size_t)32 * MOE_H;
        const _Float16* p3 = p0 + (size_t)48 * MOE_H;

        for (int k0 = 0; k0 < MOE_H; k0 += 32) {
            Frag16 a0, a1, b0, b1, b2, b3;
            a0.q[0] = *(const uint4*)(&hs[nlo][k0 + hi * 8]);
            a0.q[1] = *(const uint4*)(&hs[nlo][k0 + 16 + hi * 8]);
            a1.q[0] = *(const uint4*)(&hs[16 + nlo][k0 + hi * 8]);
            a1.q[1] = *(const uint4*)(&hs[16 + nlo][k0 + 16 + hi * 8]);
            b0.q[0] = *(const uint4*)(p0 + k0);  b0.q[1] = *(const uint4*)(p0 + k0 + 16);
            b1.q[0] = *(const uint4*)(p1 + k0);  b1.q[1] = *(const uint4*)(p1 + k0 + 16);
            b2.q[0] = *(const uint4*)(p2 + k0);  b2.q[1] = *(const uint4*)(p2 + k0 + 16);
            b3.q[0] = *(const uint4*)(p3 + k0);  b3.q[1] = *(const uint4*)(p3 + k0 + 16);
            __builtin_prefetch((const void*)(p0 + k0 + 512), 0, 3);
            __builtin_prefetch((const void*)(p2 + k0 + 512), 0, 3);
            c00 = WMMA_F16(a0, b0, c00);
            c10 = WMMA_F16(a1, b0, c10);
            c01 = WMMA_F16(a0, b1, c01);
            c11 = WMMA_F16(a1, b1, c11);
            c02 = WMMA_F16(a0, b2, c02);
            c12 = WMMA_F16(a1, b2, c12);
            c03 = WMMA_F16(a0, b3, c03);
            c13 = WMMA_F16(a1, b3, c13);
        }
    }
    // scale by gate weight, scatter-add (exactly 2 experts touch each element)
    {
        const int colB = wave * 64 + nlo;
        v8f* accs0[4] = { &c00, &c01, &c02, &c03 };
        v8f* accs1[4] = { &c10, &c11, &c12, &c13 };
#pragma unroll
        for (int nt = 0; nt < 4; ++nt) {
            const int col = colB + nt * 16;
#pragma unroll
            for (int j = 0; j < 8; ++j) {
                const int m0 = j + 8 * hi;
                const int m1 = m0 + 16;
                unsafeAtomicAdd(&out[(size_t)toks[m0] * MOE_D + col], (*accs0[nt])[j] * wgts[m0]);
                unsafeAtomicAdd(&out[(size_t)toks[m1] * MOE_D + col], (*accs1[nt])[j] * wgts[m1]);
            }
        }
    }
}

// ------------------------------------------------------------------------------
extern "C" void kernel_launch(void* const* d_in, const int* in_sizes, int n_in,
                              void* d_out, int out_size, void* d_ws, size_t ws_size,
                              hipStream_t stream) {
    const float* x  = (const float*)d_in[0];   // [B,S,D]
    const float* gw = (const float*)d_in[1];   // [D,E]
    const float* w1 = (const float*)d_in[2];   // [E,D,H]
    const float* w3 = (const float*)d_in[3];   // [E,D,H]
    const float* w2 = (const float*)d_in[4];   // [E,H,D]
    float* out = (float*)d_out;

    const int D = MOE_D, H = MOE_H, E = MOE_E;
    const int T = in_sizes[0] / D;             // 8192

    // workspace layout (≈66 MB)
    char* ws = (char*)d_ws;
    size_t o_cnt = 0;
    size_t o_tok = 256;
    size_t o_wgt = o_tok + (size_t)E * T * 4;
    size_t o_xh  = o_wgt + (size_t)E * T * 4;
    size_t o_w1t = o_xh  + (size_t)T * D * 2;
    size_t o_w3t = o_w1t + (size_t)E * H * D * 2;
    size_t o_w2t = o_w3t + (size_t)E * H * D * 2;
    int*       cnt      = (int*)(ws + o_cnt);
    int*       tok_list = (int*)(ws + o_tok);
    float*     wgt_list = (float*)(ws + o_wgt);
    _Float16*  xh       = (_Float16*)(ws + o_xh);
    _Float16*  w1t      = (_Float16*)(ws + o_w1t);
    _Float16*  w3t      = (_Float16*)(ws + o_w3t);
    _Float16*  w2t      = (_Float16*)(ws + o_w2t);

    hipMemsetAsync(cnt, 0, 256, stream);
    // residual: out = hidden_states
    hipMemcpyAsync(out, x, (size_t)T * D * sizeof(float), hipMemcpyDeviceToDevice, stream);

    // x -> f16
    {
        long long n8 = (long long)T * D / 8;
        cvt_x_kernel<<<(unsigned)((n8 + 255) / 256), 256, 0, stream>>>(x, xh, n8);
    }
    // weights -> f16, K-contiguous layouts
    transpose_cvt_kernel<<<dim3(H / 32, D / 32, E), dim3(32, 8), 0, stream>>>(w1, w1t, D, H);
    transpose_cvt_kernel<<<dim3(H / 32, D / 32, E), dim3(32, 8), 0, stream>>>(w3, w3t, D, H);
    transpose_cvt_kernel<<<dim3(D / 32, H / 32, E), dim3(32, 8), 0, stream>>>(w2, w2t, H, D);

    // router: one wave per token
    router_kernel<<<(T + 7) / 8, 256, 0, stream>>>(x, gw, cnt, tok_list, wgt_list, T);

    // expert GEMMs
    const int tiles_per_e = (T + TILE_M - 1) / TILE_M;   // 256
    moe_expert_kernel<<<E * tiles_per_e, 512, 0, stream>>>(
        xh, w1t, w3t, w2t, cnt, tok_list, wgt_list, out, T, tiles_per_e);
}
